// IntegralTransform_4990751998525
// MI455X (gfx1250) — matrix operand order repaired and checked by reading, not compiled, not run on hardware
//
#include <hip/hip_runtime.h>
#include <hip/hip_bf16.h>

typedef __attribute__((ext_vector_type(16))) _Float16 v16h;
typedef __attribute__((ext_vector_type(8)))  _Float16 v8h;
typedef __attribute__((ext_vector_type(8)))  float    v8f;

union V16H { v16h v; v8h h8[2]; };

#define WAVES_PER_BLOCK 8
#define DEG   32
#define D_F   64
#define D_HID 64

__global__ __launch_bounds__(256)
void integral_transform_wmma(const float* __restrict__ y,
                             const float* __restrict__ f_y,
                             const float* __restrict__ W0,
                             const float* __restrict__ b0,
                             const float* __restrict__ W1,
                             const float* __restrict__ b1,
                             const int*   __restrict__ nidx,
                             float* __restrict__ out,
                             int n_pts)
{
    // --- LDS layout (per 256-thread block, ~42 KB) ---
    __shared__ _Float16 lds_w1t[D_HID * D_F];                 // W1 transposed [n][k], f16 (8 KB)
    __shared__ float    lds_w0[6 * D_HID];                    // W0 [d][c] fp32
    __shared__ float    lds_b0[D_HID];
    __shared__ float    lds_b1[D_F];
    __shared__ _Float16 lds_h[WAVES_PER_BLOCK * DEG * D_HID]; // h rows per wave (32 KB)
    __shared__ int      lds_i[WAVES_PER_BLOCK * DEG];         // gathered neighbor idx

    const int tid  = threadIdx.x;
    const int lane = tid & 31;
    const int wave = tid >> 5;

    // ---- stage weights once per block ----
    for (int j = tid; j < D_HID * D_F; j += 256) {
        const int n = j >> 6;          // output channel (B column)
        const int k = j & 63;          // contraction dim
        lds_w1t[j] = (_Float16)W1[(k << 6) + n];   // transpose: [n][k]
    }
    for (int j = tid; j < 6 * D_HID; j += 256) lds_w0[j] = W0[j];
    if (tid < D_HID) lds_b0[tid] = b0[tid];
    if (tid < D_F)   lds_b1[tid] = b1[tid];
    __syncthreads();

    // one wave == one output point (32 edges, contiguous segment)
    int p = blockIdx.x * WAVES_PER_BLOCK + wave;
    if (p >= n_pts) p = n_pts - 1;     // uniform per wave; duplicate work is benign

    // ---- first layer + exact GELU, fp32: lane <-> edge ----
    const int e  = p * DEG + lane;
    const int gi = nidx[e];
    lds_i[wave * DEG + lane] = gi;

    const float r0 = y[gi * 3 + 0], r1 = y[gi * 3 + 1], r2 = y[gi * 3 + 2];
    const float s0 = y[p  * 3 + 0], s1 = y[p  * 3 + 1], s2 = y[p  * 3 + 2];

    _Float16* hrow = &lds_h[(wave * DEG + lane) * D_HID];
    for (int c = 0; c < D_HID; c += 2) {
        float t0 = lds_b0[c]
                 + r0 * lds_w0[0 * 64 + c] + r1 * lds_w0[1 * 64 + c] + r2 * lds_w0[2 * 64 + c]
                 + s0 * lds_w0[3 * 64 + c] + s1 * lds_w0[4 * 64 + c] + s2 * lds_w0[5 * 64 + c];
        float t1 = lds_b0[c + 1]
                 + r0 * lds_w0[0 * 64 + c + 1] + r1 * lds_w0[1 * 64 + c + 1] + r2 * lds_w0[2 * 64 + c + 1]
                 + s0 * lds_w0[3 * 64 + c + 1] + s1 * lds_w0[4 * 64 + c + 1] + s2 * lds_w0[5 * 64 + c + 1];
        const float g0 = 0.5f * t0 * (1.0f + erff(t0 * 0.70710678118654752f));
        const float g1 = 0.5f * t1 * (1.0f + erff(t1 * 0.70710678118654752f));
        union { _Float16 h[2]; unsigned u; } pk;
        pk.h[0] = (_Float16)g0;
        pk.h[1] = (_Float16)g1;
        *(unsigned*)(&hrow[c]) = pk.u;
    }
    __syncthreads();   // make every wave's h rows + idx visible (also orders DS ops)

    const int hs = lane >> 4;   // half-wave select (ISA A/B/C lane-group)
    const int ln = lane & 15;

    // ---- A operands: h tiles, 16x32 tiles in contiguous k order (k = 16*hs + s), matching B ----
    // lanes 0-15: K 0..15 ; lanes 16-31: K 16..31 (per 32-wide K chunk q)
    V16H A[2][2];
#pragma unroll
    for (int m = 0; m < 2; ++m)
#pragma unroll
        for (int q = 0; q < 2; ++q) {
            const _Float16* hb = &lds_h[(wave * DEG + m * 16 + ln) * D_HID + q * 32 + hs * 16];
            A[m][q].h8[0] = *(const v8h*)(hb);        // K local 16*hs .. 16*hs+7
            A[m][q].h8[1] = *(const v8h*)(hb + 8);    // K local 16*hs+8 .. 16*hs+15
        }

    // precompute gathered f_y row offsets for the 16 edge-rows this lane owns in D
    int goff[16];
#pragma unroll
    for (int m = 0; m < 2; ++m)
#pragma unroll
        for (int r = 0; r < 8; ++r)
            goff[m * 8 + r] = lds_i[wave * DEG + m * 16 + r + hs * 8] * D_F;

    // ---- 4 N-tiles: WMMA GEMM + fused (D+b1)*f_y gather + segment-sum ----
#pragma unroll
    for (int t = 0; t < 4; ++t) {
        const int col = t * 16 + ln;                       // output channel
        const _Float16* wb = &lds_w1t[col * D_HID + hs * 16];
        V16H B0, B1;
        B0.h8[0] = *(const v8h*)(wb);        // K 16*hs .. +7
        B0.h8[1] = *(const v8h*)(wb + 8);    // K 16*hs+8 .. +15
        B1.h8[0] = *(const v8h*)(wb + 32);   // K 32+16*hs ..
        B1.h8[1] = *(const v8h*)(wb + 40);

        const float bias = lds_b1[col];
        float acc = 0.0f;
#pragma unroll
        for (int m = 0; m < 2; ++m) {
            v8f cm = {};
            cm = __builtin_amdgcn_wmma_f32_16x16x32_f16(false, A[m][0].v, false, B0.v,
                                                        (short)0, cm, false, false);
            cm = __builtin_amdgcn_wmma_f32_16x16x32_f16(false, A[m][1].v, false, B1.v,
                                                        (short)0, cm, false, false);
#pragma unroll
            for (int r = 0; r < 8; ++r)
                acc += (cm[r] + bias) * f_y[goff[m * 8 + r] + col];
        }
        // partner lane (lane ^ 16) holds the complementary 16 edge-rows of this column
        acc += __shfl_xor(acc, 16, 32);
        if (lane < 16)
            out[p * D_F + col] = acc * (1.0f / 32.0f);   // mean over DEG=32 edges
    }
}

extern "C" void kernel_launch(void* const* d_in, const int* in_sizes, int n_in,
                              void* d_out, int out_size, void* d_ws, size_t ws_size,
                              hipStream_t stream) {
    const float* y    = (const float*)d_in[0];
    const float* f_y  = (const float*)d_in[1];
    const float* W0   = (const float*)d_in[2];
    const float* b0   = (const float*)d_in[3];
    const float* W1   = (const float*)d_in[4];
    const float* b1   = (const float*)d_in[5];
    const int*   nidx = (const int*)d_in[6];
    // d_in[7] = neighbors_row_splits: uniform arange*32 -> DEG fixed at 32
    const int n_pts = in_sizes[7] - 1;

    const int blocks = (n_pts + WAVES_PER_BLOCK - 1) / WAVES_PER_BLOCK;
    integral_transform_wmma<<<blocks, 256, 0, stream>>>(y, f_y, W0, b0, W1, b1, nidx,
                                                        (float*)d_out, n_pts);
}
